// Head_10926396801384
// MI455X (gfx1250) — matrix operand order repaired and checked by reading, not compiled, not run on hardware
//
#include <hip/hip_runtime.h>
#include <hip/hip_bf16.h>

#define B_ 8
#define T_ 2048
#define C_ 1024
#define H_ 128

typedef __bf16 v16bf __attribute__((ext_vector_type(16)));
typedef __bf16 v8bf  __attribute__((ext_vector_type(8)));
typedef float  v8f   __attribute__((ext_vector_type(8)));
typedef float  v4f   __attribute__((ext_vector_type(4)));
typedef unsigned int v4u __attribute__((ext_vector_type(4)));
typedef int    v4i   __attribute__((ext_vector_type(4)));
typedef int    v8i   __attribute__((ext_vector_type(8)));

#define GAS __attribute__((address_space(1)))

// ---- forced global-address-space accessors (emit global_load/store, not flat) ----
static __device__ __forceinline__ unsigned long long p2u(const void* p) {
    return __builtin_bit_cast(unsigned long long, p);
}
static __device__ __forceinline__ v4u gld_u4(const void* p) {
    return *(const GAS v4u*)p2u(p);
}
static __device__ __forceinline__ v4f gld_f4(const void* p) {
    return *(const GAS v4f*)p2u(p);
}
static __device__ __forceinline__ float gld_f(const void* p) {
    return *(const GAS float*)p2u(p);
}
static __device__ __forceinline__ int gld_i(const void* p) {
    return *(const GAS int*)p2u(p);
}
static __device__ __forceinline__ void gst_u16(void* p, unsigned short v) {
    *(GAS unsigned short*)p2u(p) = v;
}
static __device__ __forceinline__ void gst_f(void* p, float v) {
    *(GAS float*)p2u(p) = v;
}
static __device__ __forceinline__ void gst_v8bf(void* p, v8bf v) {
    *(GAS v8bf*)p2u(p) = v;
}

// native fp32 -> bf16 (hardware cvt)
static __device__ __forceinline__ __bf16 f2bf(float f) { return (__bf16)f; }

// A-matrix fragment (16x32 bf16): lane holds row; elems 0..15 = K[kb..kb+15]
// (kb = (lane>>4)*16 folded into p by caller)
static __device__ __forceinline__ v16bf load_a_g(const __bf16* p) {
    union { v16bf f; v4u u[2]; } r;
    r.u[0] = gld_u4(p);
    r.u[1] = gld_u4(p + 8);
    return r.f;
}
// B-matrix fragment (32x16 bf16): lane holds column; 16 consecutive K values
static __device__ __forceinline__ v16bf load_b_g(const __bf16* p) {
    union { v16bf f; v4u u[2]; } r;
    r.u[0] = gld_u4(p);
    r.u[1] = gld_u4(p + 8);
    return r.f;
}
// fragments from LDS (generic pointer -> ds_load_b128)
static __device__ __forceinline__ v16bf load_a_lds(const __bf16* p) {
    union { v16bf f; v4u u[2]; } r;
    r.u[0] = *(const v4u*)(p);
    r.u[1] = *(const v4u*)(p + 8);
    return r.f;
}
static __device__ __forceinline__ v16bf load_b_lds(const __bf16* p) {
    union { v16bf f; v4u u[2]; } r;
    r.u[0] = *(const v4u*)(p);
    r.u[1] = *(const v4u*)(p + 8);
    return r.f;
}

// LDS byte address of a __shared__ object (generic addr low 32 bits = LDS offset)
static __device__ __forceinline__ unsigned lds_off(const void* p) {
    return (unsigned)p2u(p);
}

// ---- Tensor Data Mover: 2-D tile (bf16) global -> LDS, descriptor per ISA §8 ----
// tile: tile1 rows x tile0 contiguous elements, row pitch stride0 (elements).
static __device__ __forceinline__ void tdm_load_2d(
    unsigned lds_addr, const void* gaddr,
    unsigned td0, unsigned td1, unsigned tile0, unsigned tile1, unsigned stride0)
{
    const unsigned long long ga = p2u(gaddr);
    v4u g0;
    g0[0] = 1u;                                               // count=1, user D#
    g0[1] = lds_addr;                                         // LDS dest (bytes)
    g0[2] = (unsigned)(ga & 0xFFFFFFFFu);                     // global_addr[31:0]
    g0[3] = (unsigned)((ga >> 32) & 0x01FFFFFFu) | 0x80000000u; // addr[56:32] | type=2
    v8i g1;
    g1[0] = (int)(1u << 16);                                  // data_size=1 (2 bytes)
    g1[1] = (int)((td0 & 0xFFFFu) << 16);                     // tensor_dim0[15:0]
    g1[2] = (int)((td0 >> 16) | ((td1 & 0xFFFFu) << 16));     // td0[31:16] | td1[15:0]
    g1[3] = (int)((td1 >> 16) | (tile0 << 16));               // td1[31:16] | tile_dim0
    g1[4] = (int)tile1;                                       // tile_dim1 (tile_dim2=0)
    g1[5] = (int)stride0;                                     // tensor_dim0_stride[31:0]
    g1[6] = 0;                                                // stride0 hi / stride1 lo
    g1[7] = 0;
    v4i z4 = {};
    v8i z8 = {};
    __builtin_amdgcn_tensor_load_to_lds(g0, g1, z4, z4, z8, 0);
}

// ---------------- Kernel 0: bulk fp32 -> bf16 conversion (streaming) ---------
__global__ __launch_bounds__(256) void cvt_bf16_kernel(
    const float* __restrict__ src, __bf16* __restrict__ dst, int n8)
{
    const int i = blockIdx.x * 256 + threadIdx.x;   // each thread: 8 elements
    if (i >= n8) return;
    v4f a = gld_f4(src + (size_t)i * 8);
    v4f b = gld_f4(src + (size_t)i * 8 + 4);
    v8bf p;
    p[0] = f2bf(a.x); p[1] = f2bf(a.y); p[2] = f2bf(a.z); p[3] = f2bf(a.w);
    p[4] = f2bf(b.x); p[5] = f2bf(b.y); p[6] = f2bf(b.z); p[7] = f2bf(b.w);
    gst_v8bf(dst + (size_t)i * 8, p);
}

// ---------------- Kernel 1: fused QKV projection (bf16 x bf16 -> f32) --------
// Wb = [3][H][C] bf16 (q,k,v concatenated). Q,K stored [B*T,H]; V stored [B,H,T].
__global__ __launch_bounds__(128) void qkv_proj_kernel(
    const __bf16* __restrict__ xb, const __bf16* __restrict__ Wb,
    const float* __restrict__ bq, const float* __restrict__ bk,
    const float* __restrict__ bv,
    __bf16* __restrict__ Qb, __bf16* __restrict__ Kb, __bf16* __restrict__ Vt)
{
    const int lane = threadIdx.x & 31;
    const int wave = threadIdx.x >> 5;
    const int hi   = lane >> 4;
    const int ln   = lane & 15;
    const int m0   = blockIdx.x * 64 + wave * 16;   // global row (B*T flattened)

    v8f acc[3][8] = {};
    const __bf16* arow = xb + (size_t)(m0 + ln) * C_ + hi * 16;

    for (int kc = 0; kc < C_ / 32; ++kc) {
        v16bf a = load_a_g(arow + kc * 32);
#pragma unroll
        for (int mat = 0; mat < 3; ++mat) {
#pragma unroll
            for (int nt = 0; nt < 8; ++nt) {
                const int n = nt * 16 + ln;
                v16bf bb = load_b_g(Wb + ((size_t)(mat * H_ + n)) * C_
                                       + kc * 32 + hi * 16);
                acc[mat][nt] = __builtin_amdgcn_wmma_f32_16x16x32_bf16(
                    false, a, false, bb, (short)0, acc[mat][nt], false, false);
            }
        }
    }

    const int b  = m0 / T_;
    const int t0 = m0 - b * T_;
    const float* bias3[3] = {bq, bk, bv};
    // Q, K: [row, h] bf16
#pragma unroll
    for (int mat = 0; mat < 2; ++mat) {
        __bf16* dst = (mat == 0) ? Qb : Kb;
#pragma unroll
        for (int nt = 0; nt < 8; ++nt) {
            const int h = nt * 16 + ln;
            const float bias = gld_f(bias3[mat] + h);
#pragma unroll
            for (int r = 0; r < 8; ++r) {
                const int row = m0 + r + 8 * hi;
                gst_u16(dst + (size_t)row * H_ + h,
                        __builtin_bit_cast(unsigned short, f2bf(acc[mat][nt][r] + bias)));
            }
        }
    }
    // V: transposed [b][h][t]; 8 consecutive t per lane -> one 16B store
#pragma unroll
    for (int nt = 0; nt < 8; ++nt) {
        const int h = nt * 16 + ln;
        const float bias = gld_f(bv + h);
        v8bf pack;
#pragma unroll
        for (int r = 0; r < 8; ++r) pack[r] = f2bf(acc[2][nt][r] + bias);
        gst_v8bf(Vt + ((size_t)(b * H_ + h)) * T_ + t0 + 8 * hi, pack);
    }
}

// ---------------- Kernel 2: flash-attention (causal, masked, TDM-fed) --------
__global__ __launch_bounds__(128) void attn_kernel(
    const __bf16* __restrict__ Qb, const __bf16* __restrict__ Kb,
    const __bf16* __restrict__ Vt, const int* __restrict__ amask,
    float* __restrict__ out)
{
    __shared__ __bf16 Klds[2][64 * 128];   // K tile: [key][h], double buffered
    __shared__ __bf16 Vlds[2][128 * 64];   // V tile: [h][t],  double buffered
    __shared__ __bf16 Plds[4][16 * 64];    // per-wave P staging (C/D -> A layout)

    const int lane = threadIdx.x & 31;
    const int wave = threadIdx.x >> 5;
    const int hi   = lane >> 4;
    const int ln   = lane & 15;
    const int b    = blockIdx.y;
    const int qb   = blockIdx.x;
    const int m0   = qb * 64 + wave * 16;    // query row within batch

    // Q fragments for this wave's 16 rows (K = H = 128 -> 4 frags) in VGPRs
    v16bf qf[4];
    {
        const __bf16* qrow = Qb + ((size_t)(b * T_) + m0 + ln) * H_;
#pragma unroll
        for (int kc = 0; kc < 4; ++kc)
            qf[kc] = load_a_g(qrow + kc * 32 + hi * 16);
    }

    v8f accO[8] = {};
    float rm[8], rs[8];
#pragma unroll
    for (int r = 0; r < 8; ++r) { rm[r] = -1e30f; rs[r] = 0.0f; }
    const float scale = 0.08838834764831845f;   // 1/sqrt(128)

    // prologue: TDM-issue tiles for kb = 0 into buffer 0 (one wave drives DMA)
    if (wave == 0) {
        tdm_load_2d(lds_off(&Klds[0][0]), Kb + (size_t)(b * T_) * H_,
                    H_, 64, H_, 64, H_);
        tdm_load_2d(lds_off(&Vlds[0][0]), Vt + (size_t)(b * H_) * T_,
                    64, H_, 64, H_, T_);
    }

    for (int kb = 0; kb <= qb; ++kb) {
        const int buf = kb & 1;
        if (wave == 0) __builtin_amdgcn_s_wait_tensorcnt(0);
        __syncthreads();   // tile kb visible; all waves done with buffer buf^1
        if (wave == 0 && kb < qb) {   // overlap DMA of kb+1 with compute of kb
            tdm_load_2d(lds_off(&Klds[buf ^ 1][0]),
                        Kb + ((size_t)(b * T_) + (kb + 1) * 64) * H_,
                        H_, 64, H_, 64, H_);
            tdm_load_2d(lds_off(&Vlds[buf ^ 1][0]),
                        Vt + (size_t)(b * H_) * T_ + (kb + 1) * 64,
                        64, H_, 64, H_, T_);
        }

        // ---- S = Q * K^T  (16 x 64), 16 WMMAs, K tile from LDS ----
        v8f s[4] = {};
#pragma unroll
        for (int nt = 0; nt < 4; ++nt) {
            const __bf16* krow = &Klds[buf][(nt * 16 + ln) * H_];
#pragma unroll
            for (int kc = 0; kc < 4; ++kc) {
                v16bf bb = load_b_lds(krow + kc * 32 + hi * 16);
                s[nt] = __builtin_amdgcn_wmma_f32_16x16x32_bf16(
                    false, qf[kc], false, bb, (short)0, s[nt], false, false);
            }
        }
        // ---- scale + causal & attention mask (branchless; EXEC stays full) ----
#pragma unroll
        for (int nt = 0; nt < 4; ++nt) {
            const int col = kb * 64 + nt * 16 + ln;
            const bool ok = gld_i(amask + b * T_ + col) != 0;
#pragma unroll
            for (int r = 0; r < 8; ++r) {
                const int row = m0 + r + 8 * hi;
                float v = s[nt][r] * scale;
                s[nt][r] = (ok && col <= row) ? v : -__builtin_inff();
            }
        }
        // ---- online softmax (row reductions across 16-lane halves) ----
#pragma unroll
        for (int r = 0; r < 8; ++r) {
            float m = fmaxf(fmaxf(s[0][r], s[1][r]), fmaxf(s[2][r], s[3][r]));
#pragma unroll
            for (int off = 1; off < 16; off <<= 1) m = fmaxf(m, __shfl_xor(m, off, 32));
            const float mnew  = fmaxf(rm[r], m);
            const float alpha = __expf(rm[r] - mnew);
            float psum = 0.0f;
#pragma unroll
            for (int nt = 0; nt < 4; ++nt) {
                float p = __expf(s[nt][r] - mnew);
                s[nt][r] = p;
                psum += p;
            }
#pragma unroll
            for (int off = 1; off < 16; off <<= 1) psum += __shfl_xor(psum, off, 32);
            rs[r] = rs[r] * alpha + psum;
            rm[r] = mnew;
#pragma unroll
            for (int nt = 0; nt < 8; ++nt) accO[nt][r] *= alpha;
        }
        // ---- P: C/D layout -> LDS row-major -> A layout ----
        __bf16* pl = Plds[wave];
#pragma unroll
        for (int nt = 0; nt < 4; ++nt) {
            const int c = nt * 16 + ln;
#pragma unroll
            for (int r = 0; r < 8; ++r)
                pl[(r + 8 * hi) * 64 + c] = f2bf(s[nt][r]);
        }
        // ---- O += P * V  (K = 64 keys -> 2 steps, 16 WMMAs, V tile from LDS) ----
#pragma unroll
        for (int kk = 0; kk < 2; ++kk) {
            v16bf pa = load_a_lds(pl + ln * 64 + kk * 32 + hi * 16);
#pragma unroll
            for (int nt = 0; nt < 8; ++nt) {
                const __bf16* vrow = &Vlds[buf][(nt * 16 + ln) * 64];
                v16bf bb = load_b_lds(vrow + kk * 32 + hi * 16);
                accO[nt] = __builtin_amdgcn_wmma_f32_16x16x32_bf16(
                    false, pa, false, bb, (short)0, accO[nt], false, false);
            }
        }
    }

    // ---- normalize + store fp32 output [B,T,H] ----
#pragma unroll
    for (int nt = 0; nt < 8; ++nt) {
        const int h = nt * 16 + ln;
#pragma unroll
        for (int r = 0; r < 8; ++r) {
            const int row = m0 + r + 8 * hi;
            gst_f(out + ((size_t)(b * T_) + row) * H_ + h, accO[nt][r] / rs[r]);
        }
    }
}

extern "C" void kernel_launch(void* const* d_in, const int* in_sizes, int n_in,
                              void* d_out, int out_size, void* d_ws, size_t ws_size,
                              hipStream_t stream) {
    const float* x     = (const float*)d_in[0];
    const int*   amask = (const int*)  d_in[1];
    const float* Wq    = (const float*)d_in[2];
    const float* bq    = (const float*)d_in[3];
    const float* Wk    = (const float*)d_in[4];
    const float* bk    = (const float*)d_in[5];
    const float* Wv    = (const float*)d_in[6];
    const float* bv    = (const float*)d_in[7];
    float* out = (float*)d_out;

    const size_t qkv = (size_t)B_ * T_ * H_;      // 2,097,152 elems
    const size_t xel = (size_t)B_ * T_ * C_;      // 16,777,216 elems
    const size_t wel = (size_t)H_ * C_;           // 131,072 elems
    __bf16* Qb = (__bf16*)d_ws;                   //  4 MB
    __bf16* Kb = Qb + qkv;                        //  4 MB
    __bf16* Vt = Kb + qkv;                        //  4 MB  ([B][H][T])
    __bf16* xb = Vt + qkv;                        // 32 MB
    __bf16* Wb = xb + xel;                        // 0.75 MB ([3][H][C])

    // one-pass fp32 -> bf16 conversions (bandwidth bound)
    cvt_bf16_kernel<<<dim3((int)(xel / 8 / 256)), dim3(256), 0, stream>>>(x, xb, (int)(xel / 8));
    cvt_bf16_kernel<<<dim3((int)(wel / 8 / 256)), dim3(256), 0, stream>>>(Wq, Wb,            (int)(wel / 8));
    cvt_bf16_kernel<<<dim3((int)(wel / 8 / 256)), dim3(256), 0, stream>>>(Wk, Wb + wel,      (int)(wel / 8));
    cvt_bf16_kernel<<<dim3((int)(wel / 8 / 256)), dim3(256), 0, stream>>>(Wv, Wb + 2 * wel,  (int)(wel / 8));

    qkv_proj_kernel<<<dim3((B_ * T_) / 64), dim3(128), 0, stream>>>(
        xb, Wb, bq, bk, bv, Qb, Kb, Vt);
    attn_kernel<<<dim3(T_ / 64, B_), dim3(128), 0, stream>>>(
        Qb, Kb, Vt, amask, out);
}
